// GraphTransformerLayer_63814624084399
// MI455X (gfx1250) — matrix operand order repaired and checked by reading, not compiled, not run on hardware
//
#include <hip/hip_runtime.h>

// GraphTransformer layer for MI455X (gfx1250, wave32).
// N nodes, E edges, D=256, H=8 heads, C=32 channels/head.

typedef __attribute__((ext_vector_type(16))) __bf16 v16bf;
typedef __attribute__((ext_vector_type(8)))  float  v8f;

__device__ inline unsigned short f32_to_bf16(float f) {
    unsigned u = __float_as_uint(f);
    u += 0x7FFFu + ((u >> 16) & 1u);   // round-to-nearest-even
    return (unsigned short)(u >> 16);
}
__device__ inline float bf16u_lo(unsigned u) { return __uint_as_float(u << 16); }
__device__ inline float bf16u_hi(unsigned u) { return __uint_as_float(u & 0xffff0000u); }

__device__ inline void atomicMaxF(float* addr, float v) {
    if (v >= 0.0f) atomicMax((int*)addr, __float_as_int(v));
    else           atomicMin((unsigned int*)addr, __float_as_uint(v));
}

union FragBF16 { uint4 u[2]; v16bf v; };

// ---------------------------------------------------------------------------
// Pre-pack a 256x256 f32 weight matrix into bf16 WMMA B-fragment order:
//   Wp[((ks*16 + nt)*32 + lane)*16 + e]  (bf16)
// holds W[(ks*32 + (lane>=16 ? 16:0) + e)*256 + nt*16 + (lane&15)].
// ---------------------------------------------------------------------------
__global__ __launch_bounds__(256)
void pack_w_kernel(const float* __restrict__ W, unsigned short* __restrict__ out)
{
    int idx = blockIdx.x * blockDim.x + threadIdx.x;
    if (idx >= 8 * 16 * 32) return;
    int lane = idx & 31;
    int nt   = (idx >> 5) & 15;
    int ks   = idx >> 9;
    int n     = nt * 16 + (lane & 15);
    int kbase = ks * 32 + (lane >> 4) * 16;

    unsigned short tmp[16];
    #pragma unroll
    for (int e = 0; e < 16; ++e)
        tmp[e] = f32_to_bf16(W[(size_t)(kbase + e) * 256 + n]);

    uint4* dst = (uint4*)(out + (size_t)idx * 16);
    dst[0] = *(const uint4*)&tmp[0];
    dst[1] = *(const uint4*)&tmp[8];
}

// ---------------------------------------------------------------------------
// Tiled GEMM: out[M x 256] = A[M x 256] @ W[256 x 256] (+ bias), bf16 WMMA.
// Block = 256 threads (8 waves). Block tile = 64 rows x 256 cols.
// Wave w: row group w&3 (16 rows), col group w>>2 (128 cols = 8 WMMA n-tiles).
// Pipelining: A tile for ks+1 loaded to VGPRs during ks's WMMAs; all 8 B
// fragments of a K-step issued as one load clause before the 8 WMMAs.
// ---------------------------------------------------------------------------
template <bool STORE_BF16>
__global__ __launch_bounds__(256)
void gemm256_core(const float* __restrict__ A,
                  const unsigned short* __restrict__ Wp,  // packed bf16 fragments
                  const float* __restrict__ bias,
                  void* __restrict__ outv, int M)
{
    // A tile in fragment order: [rg(4)][lane(32)][elem(16)] bf16 = 4KB
    __shared__ unsigned short ldsA[4 * 32 * 16];

    const int tid  = threadIdx.x;
    const int wave = tid >> 5;
    const int lane = tid & 31;
    const int half = lane >> 4;
    const int lr   = lane & 15;
    const int rg   = wave & 3;      // row group
    const int cg   = wave >> 2;     // col group
    const int row0 = blockIdx.x * 64;

    // staging coordinates (fixed per thread)
    const int sr    = tid >> 2;            // tile row 0..63
    const int scb   = (tid & 3) * 8;       // k sub-block 0,8,16,24
    const int shalf = scb >> 4;
    const int sebase= ((scb >> 3) & 1) * 8; // elem base 0 or 8
    const int sslot = ((sr >> 4) * 32 + (sr & 15) + 16 * shalf) * 16 + sebase;
    const bool sin  = (row0 + sr) < M;
    const float* arow = A + (size_t)(row0 + sr) * 256 + scb;

    v8f acc[8];
    #pragma unroll
    for (int i = 0; i < 8; ++i)
        #pragma unroll
        for (int j = 0; j < 8; ++j) acc[i][j] = 0.0f;

    // prologue: A data for ks=0
    float4 f0 = make_float4(0.f, 0.f, 0.f, 0.f), f1 = f0;
    if (sin) {
        f0 = *(const float4*)arow;
        f1 = *(const float4*)(arow + 4);
    }

    for (int ks = 0; ks < 8; ++ks) {
        // ---- convert staged A data, one ds_store_b128 per thread ----
        {
            unsigned short us[8];
            us[0] = f32_to_bf16(f0.x); us[1] = f32_to_bf16(f0.y);
            us[2] = f32_to_bf16(f0.z); us[3] = f32_to_bf16(f0.w);
            us[4] = f32_to_bf16(f1.x); us[5] = f32_to_bf16(f1.y);
            us[6] = f32_to_bf16(f1.z); us[7] = f32_to_bf16(f1.w);
            *(uint4*)&ldsA[sslot] = *(const uint4*)us;
        }
        __syncthreads();

        // ---- pipeline: issue next K-step's A loads (latency hidden by WMMAs) ----
        if (ks + 1 < 8 && sin) {
            const float* ap = arow + (ks + 1) * 32;
            f0 = *(const float4*)ap;
            f1 = *(const float4*)(ap + 4);
            if (ks + 2 < 8) __builtin_prefetch(ap + 32, 0, 1);
        }

        // ---- A fragment: 2 x ds_load_b128 (lane's 32B slot) ----
        FragBF16 afr;
        {
            const uint4* ap128 = (const uint4*)&ldsA[(rg * 32 + lane) * 16];
            afr.u[0] = ap128[0];
            afr.u[1] = ap128[1];
        }
        // ---- issue all 8 B fragments as one clause, then 8 WMMAs ----
        const uint4* bbase = (const uint4*)(Wp +
            ((((size_t)ks * 16 + cg * 8) * 32 + lane) << 4));
        FragBF16 bfr[8];
        #pragma unroll
        for (int nt = 0; nt < 8; ++nt) {
            bfr[nt].u[0] = bbase[nt * 64];       // each n-tile block = 1024B = 64 uint4
            bfr[nt].u[1] = bbase[nt * 64 + 1];
        }
        #pragma unroll
        for (int nt = 0; nt < 8; ++nt) {
            acc[nt] = __builtin_amdgcn_wmma_f32_16x16x32_bf16(
                false, afr.v, false, bfr[nt].v, (short)0, acc[nt], false, false);
        }
        __syncthreads();
    }

    // epilogue: C/D layout vgpr r -> M = r (lanes 0-15) / 8+r (lanes 16-31), N = lane&15
    #pragma unroll
    for (int nt = 0; nt < 8; ++nt) {
        int n = cg * 128 + nt * 16 + lr;
        float bi = bias ? bias[n] : 0.0f;
        #pragma unroll
        for (int r = 0; r < 8; ++r) {
            int m = row0 + rg * 16 + r + half * 8;
            if (m < M) {
                float val = acc[nt][r] + bi;
                if (STORE_BF16)
                    ((unsigned short*)outv)[(size_t)m * 256 + n] = f32_to_bf16(val);
                else
                    ((float*)outv)[(size_t)m * 256 + n] = val;
            }
        }
    }
}

// ---------------------------------------------------------------------------
__global__ void init_stats_kernel(float* __restrict__ amax,
                                  float* __restrict__ denom, int n)
{
    int i = blockIdx.x * blockDim.x + threadIdx.x;
    if (i < n) { amax[i] = -__builtin_inff(); denom[i] = 0.0f; }
}

// One wave per edge: alpha[e][h] = <q[dst,h], k[src,h]+e[e,h]> / sqrt(C)
__global__ __launch_bounds__(256)
void edge_alpha_kernel(const int* __restrict__ ei, const float* __restrict__ q,
                       const float* __restrict__ k, const unsigned short* __restrict__ e16,
                       float* __restrict__ alpha, float* __restrict__ amax, int E)
{
    int edge = blockIdx.x * 8 + (threadIdx.x >> 5);
    if (edge >= E) return;
    int lane = threadIdx.x & 31;
    int src = ei[edge];
    int dst = ei[E + edge];
    int c0 = lane * 8;                     // lane covers 8 channels; head = lane>>2

    const float4* qp = (const float4*)(q + (size_t)dst * 256 + c0);
    const float4* kp = (const float4*)(k + (size_t)src * 256 + c0);
    const uint4*  ep = (const uint4*)(e16 + (size_t)edge * 256 + c0);
    float4 q0 = qp[0], q1 = qp[1];
    float4 k0 = kp[0], k1 = kp[1];
    uint4  eu = ep[0];

    float p = 0.0f;
    p += q0.x * (k0.x + bf16u_lo(eu.x));
    p += q0.y * (k0.y + bf16u_hi(eu.x));
    p += q0.z * (k0.z + bf16u_lo(eu.y));
    p += q0.w * (k0.w + bf16u_hi(eu.y));
    p += q1.x * (k1.x + bf16u_lo(eu.z));
    p += q1.y * (k1.y + bf16u_hi(eu.z));
    p += q1.z * (k1.z + bf16u_lo(eu.w));
    p += q1.w * (k1.w + bf16u_hi(eu.w));

    p += __shfl_xor(p, 1, 32);
    p += __shfl_xor(p, 2, 32);             // lanes 4h now hold head-h dot product
    if ((lane & 3) == 0) {
        int h = lane >> 2;
        float a = p * 0.17677669529663689f;  // 1/sqrt(32)
        alpha[(size_t)edge * 8 + h] = a;
        atomicMaxF(&amax[(size_t)dst * 8 + h], a);
    }
}

// One thread per (edge, head): ea = exp(alpha - amax[dst]); denom += ea
__global__ __launch_bounds__(256)
void edge_exp_kernel(const int* __restrict__ ei, float* __restrict__ alpha,
                     const float* __restrict__ amax, float* __restrict__ denom, int E)
{
    int idx = blockIdx.x * blockDim.x + threadIdx.x;
    if (idx >= E * 8) return;
    int e = idx >> 3, h = idx & 7;
    int dst = ei[E + e];
    float ea = __expf(alpha[idx] - amax[(size_t)dst * 8 + h]);
    alpha[idx] = ea;
    atomicAdd(&denom[(size_t)dst * 8 + h], ea);
}

// One wave per edge: out[dst] += (v[src]+e) * ea/(denom[dst]+1e-16)
__global__ __launch_bounds__(256)
void edge_scatter_kernel(const int* __restrict__ ei, const float* __restrict__ v,
                         const unsigned short* __restrict__ e16,
                         const float* __restrict__ alpha, const float* __restrict__ denom,
                         float* __restrict__ out, int E)
{
    int edge = blockIdx.x * 8 + (threadIdx.x >> 5);
    if (edge >= E) return;
    int lane = threadIdx.x & 31;
    int src = ei[edge];
    int dst = ei[E + edge];
    int h  = lane >> 2;
    int c0 = lane * 8;

    float w = alpha[(size_t)edge * 8 + h] / (denom[(size_t)dst * 8 + h] + 1e-16f);

    const float4* vp = (const float4*)(v + (size_t)src * 256 + c0);
    const uint4*  ep = (const uint4*)(e16 + (size_t)edge * 256 + c0);
    float4 v0 = vp[0], v1 = vp[1];
    uint4  eu = ep[0];

    float m[8];
    m[0] = (v0.x + bf16u_lo(eu.x)) * w;
    m[1] = (v0.y + bf16u_hi(eu.x)) * w;
    m[2] = (v0.z + bf16u_lo(eu.y)) * w;
    m[3] = (v0.w + bf16u_hi(eu.y)) * w;
    m[4] = (v1.x + bf16u_lo(eu.z)) * w;
    m[5] = (v1.y + bf16u_hi(eu.z)) * w;
    m[6] = (v1.z + bf16u_lo(eu.w)) * w;
    m[7] = (v1.w + bf16u_hi(eu.w)) * w;

    float* op = out + (size_t)dst * 256 + c0;
    #pragma unroll
    for (int j = 0; j < 8; ++j) atomicAdd(op + j, m[j]);
}

// ---------------------------------------------------------------------------
extern "C" void kernel_launch(void* const* d_in, const int* in_sizes, int n_in,
                              void* d_out, int out_size, void* d_ws, size_t ws_size,
                              hipStream_t stream)
{
    (void)n_in; (void)out_size; (void)ws_size;
    const float* x     = (const float*)d_in[0];
    const int*   ei    = (const int*)  d_in[1];
    const float* eattr = (const float*)d_in[2];
    const float* Wq = (const float*)d_in[3];
    const float* bq = (const float*)d_in[4];
    const float* Wk = (const float*)d_in[5];
    const float* bk = (const float*)d_in[6];
    const float* Wv = (const float*)d_in[7];
    const float* bv = (const float*)d_in[8];
    const float* We = (const float*)d_in[9];
    const float* Ws = (const float*)d_in[10];
    const float* bs = (const float*)d_in[11];

    const int N = in_sizes[0] / 256;
    const int E = in_sizes[2] / 256;

    // workspace carve-out (256B aligned slabs)
    char* p = (char*)d_ws;
    auto take = [&](size_t bytes) -> void* {
        void* r = (void*)p;
        p += (bytes + 255) & ~(size_t)255;
        return r;
    };
    float* q     = (float*)take((size_t)N * 256 * sizeof(float));
    float* k     = (float*)take((size_t)N * 256 * sizeof(float));
    float* v     = (float*)take((size_t)N * 256 * sizeof(float));
    float* amax  = (float*)take((size_t)N * 8 * sizeof(float));
    float* denom = (float*)take((size_t)N * 8 * sizeof(float));
    float* alpha = (float*)take((size_t)E * 8 * sizeof(float));
    unsigned short* ebf = (unsigned short*)take((size_t)E * 256 * sizeof(unsigned short));
    const size_t WPK = (size_t)8 * 16 * 32 * 16;  // packed-W elements (bf16)
    unsigned short* WqP = (unsigned short*)take(WPK * 2);
    unsigned short* WkP = (unsigned short*)take(WPK * 2);
    unsigned short* WvP = (unsigned short*)take(WPK * 2);
    unsigned short* WsP = (unsigned short*)take(WPK * 2);
    unsigned short* WeP = (unsigned short*)take(WPK * 2);

    // pre-pack weights into WMMA B-fragment order (bf16)
    pack_w_kernel<<<16, 256, 0, stream>>>(Wq, WqP);
    pack_w_kernel<<<16, 256, 0, stream>>>(Wk, WkP);
    pack_w_kernel<<<16, 256, 0, stream>>>(Wv, WvP);
    pack_w_kernel<<<16, 256, 0, stream>>>(Ws, WsP);
    pack_w_kernel<<<16, 256, 0, stream>>>(We, WeP);

    const int nh = N * 8;
    init_stats_kernel<<<(nh + 255) / 256, 256, 0, stream>>>(amax, denom, nh);

    const int gN = (N + 63) / 64;
    const int gE = (E + 63) / 64;
    gemm256_core<false><<<gN, 256, 0, stream>>>(x, WqP, bq, (void*)q, N);
    gemm256_core<false><<<gN, 256, 0, stream>>>(x, WkP, bk, (void*)k, N);
    gemm256_core<false><<<gN, 256, 0, stream>>>(x, WvP, bv, (void*)v, N);
    gemm256_core<false><<<gN, 256, 0, stream>>>(x, WsP, bs, d_out, N);       // skip path
    gemm256_core<true ><<<gE, 256, 0, stream>>>(eattr, WeP, nullptr, (void*)ebf, E);

    edge_alpha_kernel<<<(E + 7) / 8, 256, 0, stream>>>(ei, q, k, ebf, alpha, amax, E);
    edge_exp_kernel<<<(E * 8 + 255) / 256, 256, 0, stream>>>(ei, alpha, amax, denom, E);
    edge_scatter_kernel<<<(E + 7) / 8, 256, 0, stream>>>(ei, v, ebf, alpha, denom,
                                                         (float*)d_out, E);
}
